// Model_65687229825820
// MI455X (gfx1250) — compile-verified, matrix operand order checked by reading, not run on hardware
//
#include <hip/hip_runtime.h>
#include <hip/hip_bf16.h>
#include <math.h>

// Problem constants (match reference).
#define SEQ_N 512
#define DIM   1024
#define VOCAB 2048
// BOS=0, EOS=1

typedef __attribute__((ext_vector_type(16))) __bf16 v16bf;
typedef __attribute__((ext_vector_type(8)))  float  v8f;

// float -> bf16 bits (RNE), low 16 bits of result valid.
static __device__ __forceinline__ unsigned f2bfu(float f) {
  unsigned u = __float_as_uint(f);
  return ((u + 0x7FFFu + ((u >> 16) & 1u)) >> 16) & 0xFFFFu;
}

union FragU {
  v16bf v;
  uint4 q[2];
};

// ---------------------------------------------------------------------------
// Fragment packing. WMMA bf16 16x16x32 lane layout (ISA 7.12.2):
//   lane&15 -> M (A) or N (B); lane>>4 -> +8 K offset;
//   VGPR v (0..7) holds K pair k(v) = (v&3)*2 + (v>>2)*16 + (lane>>4)*8.
// Packed element layout (uint4 = 4 bf16 pairs = VGPRs h*4..h*4+3):
//   P[(((tile * KB + kb) * 2 + h) * 32 + lane)]
// so a wave fetches a whole fragment as two coalesced b128 loads.
// ---------------------------------------------------------------------------

// A' row m = A[gather[m]] (or A[m]); A row-major [M,K] -> pairs contiguous.
__global__ __launch_bounds__(256) void pack_a_bf16(
    const float* __restrict__ A, const int* __restrict__ gather,
    uint4* __restrict__ Ap, int M, int K)
{
  const int KB  = K >> 5;
  const int idx = blockIdx.x * 256 + threadIdx.x;
  if (idx >= (M >> 4) * KB * 64) return;
  const int lane = idx & 31;
  const int h    = (idx >> 5) & 1;
  const int kb   = (idx >> 6) % KB;
  const int mt   = (idx >> 6) / KB;

  const int row  = (mt << 4) + (lane & 15);
  const int srow = gather ? gather[row] : row;
  const float* __restrict__ ar = A + (size_t)srow * K;
  const int halfK = (lane >> 4) << 3;

  uint4 o;
  unsigned d[4];
#pragma unroll
  for (int vv = 0; vv < 4; ++vv) {
    const int v = (h << 2) + vv;
    const int k = (kb << 5) + ((v & 3) << 1) + ((v >> 2) << 4) + halfK;
    const float2 x = *(const float2*)(ar + k);
    d[vv] = f2bfu(x.x) | (f2bfu(x.y) << 16);
  }
  o.x = d[0]; o.y = d[1]; o.z = d[2]; o.w = d[3];
  Ap[idx] = o;
}

// W row-major [K,N]; element (k, col) at W[k*N+col] (pair strided by N).
__global__ __launch_bounds__(256) void pack_b_bf16(
    const float* __restrict__ W, uint4* __restrict__ Bp, int N, int K)
{
  const int KB  = K >> 5;
  const int idx = blockIdx.x * 256 + threadIdx.x;
  if (idx >= (N >> 4) * KB * 64) return;
  const int lane = idx & 31;
  const int h    = (idx >> 5) & 1;
  const int kb   = (idx >> 6) % KB;
  const int nt   = (idx >> 6) / KB;

  const int col   = (nt << 4) + (lane & 15);
  const int halfK = (lane >> 4) << 3;

  uint4 o;
  unsigned d[4];
#pragma unroll
  for (int vv = 0; vv < 4; ++vv) {
    const int v = (h << 2) + vv;
    const int k = (kb << 5) + ((v & 3) << 1) + ((v >> 2) << 4) + halfK;
    d[vv] = f2bfu(W[(size_t)k * N + col]) |
            (f2bfu(W[(size_t)(k + 1) * N + col]) << 16);
  }
  o.x = d[0]; o.y = d[1]; o.z = d[2]; o.w = d[3];
  Bp[idx] = o;
}

// ---------------------------------------------------------------------------
// GEMM on pre-packed bf16 fragments: C[M,N] = A@W + bias, fp32 accumulate.
// One wave computes a 16x64 strip: 1 A fragment reused by 4 B fragments,
// 4 v_wmma_f32_16x16x32_bf16 per K-step. Register double-buffering: the
// fragment set for K-step k+1 is in flight while step k's WMMAs issue, so
// the per-WMMA s_wait_loadcnt covers loads issued ~4 WMMAs earlier.
// ---------------------------------------------------------------------------
__global__ __launch_bounds__(256) void wmma_gemm_packed(
    const uint4* __restrict__ Ap, const uint4* __restrict__ Bp,
    const float* __restrict__ bias, float* __restrict__ C,
    int M, int N, int K)
{
  const int lane    = threadIdx.x & 31;
  const int wave    = threadIdx.x >> 5;
  const int KB      = K >> 5;            // even (K multiple of 64)
  const int nstrips = N >> 6;
  const int strip   = blockIdx.x * 8 + wave;
  if (strip >= (M >> 4) * nstrips) return;      // wave-uniform guard
  const int ms = strip / nstrips;
  const int ns = strip % nstrips;
  const int m0 = ms << 4;
  const int n0 = ns << 6;

  const uint4* __restrict__ ap  = Ap + (size_t)ms * KB * 64 + lane;
  const uint4* __restrict__ bp0 = Bp + (size_t)(ns * 4 + 0) * KB * 64 + lane;
  const uint4* __restrict__ bp1 = Bp + (size_t)(ns * 4 + 1) * KB * 64 + lane;
  const uint4* __restrict__ bp2 = Bp + (size_t)(ns * 4 + 2) * KB * 64 + lane;
  const uint4* __restrict__ bp3 = Bp + (size_t)(ns * 4 + 3) * KB * 64 + lane;

  v8f acc0 = {}, acc1 = {}, acc2 = {}, acc3 = {};
  FragU aA, b0A, b1A, b2A, b3A;   // set A
  FragU aB, b0B, b1B, b2B, b3B;   // set B

#define LOAD_SET(sa, s0, s1, s2, s3)                         \
  do {                                                       \
    sa.q[0] = ap[0];  sa.q[1] = ap[32];  ap  += 64;          \
    s0.q[0] = bp0[0]; s0.q[1] = bp0[32]; bp0 += 64;          \
    s1.q[0] = bp1[0]; s1.q[1] = bp1[32]; bp1 += 64;          \
    s2.q[0] = bp2[0]; s2.q[1] = bp2[32]; bp2 += 64;          \
    s3.q[0] = bp3[0]; s3.q[1] = bp3[32]; bp3 += 64;          \
  } while (0)

#define WMMA_SET(sa, s0, s1, s2, s3)                                          \
  do {                                                                        \
    acc0 = __builtin_amdgcn_wmma_f32_16x16x32_bf16(false, sa.v, false, s0.v,  \
                                                   (short)0, acc0, false, false); \
    acc1 = __builtin_amdgcn_wmma_f32_16x16x32_bf16(false, sa.v, false, s1.v,  \
                                                   (short)0, acc1, false, false); \
    acc2 = __builtin_amdgcn_wmma_f32_16x16x32_bf16(false, sa.v, false, s2.v,  \
                                                   (short)0, acc2, false, false); \
    acc3 = __builtin_amdgcn_wmma_f32_16x16x32_bf16(false, sa.v, false, s3.v,  \
                                                   (short)0, acc3, false, false); \
  } while (0)

  LOAD_SET(aA, b0A, b1A, b2A, b3A);                 // kb = 0
  for (int kb = 0; kb < KB - 2; kb += 2) {
    LOAD_SET(aB, b0B, b1B, b2B, b3B);               // kb+1 in flight
    WMMA_SET(aA, b0A, b1A, b2A, b3A);               // consume kb
    LOAD_SET(aA, b0A, b1A, b2A, b3A);               // kb+2 in flight
    WMMA_SET(aB, b0B, b1B, b2B, b3B);               // consume kb+1
  }
  LOAD_SET(aB, b0B, b1B, b2B, b3B);                 // kb = KB-1
  WMMA_SET(aA, b0A, b1A, b2A, b3A);                 // consume KB-2
  WMMA_SET(aB, b0B, b1B, b2B, b3B);                 // consume KB-1

#undef LOAD_SET
#undef WMMA_SET

  // C/D layout: lane&15 = col, VGPR r = row (lane>>4)*8 + r.
  const int   coll  = lane & 15;
  const int   rbase = (lane >> 4) << 3;
  const float bv0 = bias[n0 + coll];
  const float bv1 = bias[n0 + 16 + coll];
  const float bv2 = bias[n0 + 32 + coll];
  const float bv3 = bias[n0 + 48 + coll];
#pragma unroll
  for (int r = 0; r < 8; ++r) {
    float* crow = C + (size_t)(m0 + rbase + r) * N + n0 + coll;
    crow[0]  = acc0[r] + bv0;
    crow[16] = acc1[r] + bv1;
    crow[32] = acc2[r] + bv2;
    crow[48] = acc3[r] + bv3;
  }
}

// ---------------------------------------------------------------------------
// Elman scan: h_t = tanh(XW[t] + h_{t-1} @ Wh). One persistent workgroup,
// 256 threads x 4 columns. h in LDS; Wh (4 MB) streams from L2.
// ---------------------------------------------------------------------------
__global__ __launch_bounds__(256) void rnn_scan(
    const float* __restrict__ XW, const float* __restrict__ Wh,
    float* __restrict__ Hout)
{
  __shared__ float h[DIM];
  const int j4 = (int)threadIdx.x << 2;
  *(float4*)&h[j4] = make_float4(0.f, 0.f, 0.f, 0.f);
  __syncthreads();

  for (int t = 0; t < SEQ_N; ++t) {
    float4 acc = *(const float4*)(XW + (size_t)t * DIM + j4);   // bias folded in
    for (int k = 0; k < DIM; k += 4) {
      const float4 hv = *(const float4*)&h[k];                  // ds_load_b128
      const float4 w0 = *(const float4*)(Wh + (size_t)(k + 0) * DIM + j4);
      const float4 w1 = *(const float4*)(Wh + (size_t)(k + 1) * DIM + j4);
      const float4 w2 = *(const float4*)(Wh + (size_t)(k + 2) * DIM + j4);
      const float4 w3 = *(const float4*)(Wh + (size_t)(k + 3) * DIM + j4);
      __builtin_prefetch(Wh + (size_t)((k + 64) & (DIM - 1)) * DIM + j4, 0, 0);
      acc.x = fmaf(hv.x, w0.x, fmaf(hv.y, w1.x, fmaf(hv.z, w2.x, fmaf(hv.w, w3.x, acc.x))));
      acc.y = fmaf(hv.x, w0.y, fmaf(hv.y, w1.y, fmaf(hv.z, w2.y, fmaf(hv.w, w3.y, acc.y))));
      acc.z = fmaf(hv.x, w0.z, fmaf(hv.y, w1.z, fmaf(hv.z, w2.z, fmaf(hv.w, w3.z, acc.z))));
      acc.w = fmaf(hv.x, w0.w, fmaf(hv.y, w1.w, fmaf(hv.z, w2.w, fmaf(hv.w, w3.w, acc.w))));
    }
    float4 hn;
    hn.x = tanhf(acc.x); hn.y = tanhf(acc.y);
    hn.z = tanhf(acc.z); hn.w = tanhf(acc.w);
    __syncthreads();                 // everyone done reading h
    *(float4*)&h[j4] = hn;
    *(float4*)(Hout + (size_t)t * DIM + j4) = hn;
    __syncthreads();                 // new h visible
  }
}

// ---------------------------------------------------------------------------
// expTt[j*V + r] = exp(T[r*V + j])  (transposed so the CRF matvec is
// coalesced over output rows). 32x32 LDS tiles.
// ---------------------------------------------------------------------------
__global__ __launch_bounds__(256) void exp_transpose(
    const float* __restrict__ T, float* __restrict__ Tt)
{
  __shared__ float tile[32][33];
  const int bx = (int)blockIdx.x << 5;   // source col j
  const int by = (int)blockIdx.y << 5;   // source row r
  const int x  = threadIdx.x;
  for (int yy = threadIdx.y; yy < 32; yy += 8)
    tile[yy][x] = __expf(T[(size_t)(by + yy) * VOCAB + bx + x]);
  __syncthreads();
  for (int yy = threadIdx.y; yy < 32; yy += 8)
    Tt[(size_t)(bx + yy) * VOCAB + by + x] = tile[x][yy];
}

// ---------------------------------------------------------------------------
// CRF backward scan in exp domain:
//   m = max(c); w = exp(c - m); s = expT^T w; c' = O[t] + m + log(s)
// One persistent workgroup, 1024 threads x 2 rows (held in registers).
// ---------------------------------------------------------------------------
__global__ __launch_bounds__(1024) void crf_scan(
    const float* __restrict__ O, const float* __restrict__ expTt,
    float* __restrict__ out)
{
  __shared__ float red[1024];
  __shared__ float wbuf[VOCAB];
  const int tid = threadIdx.x;
  const int r0  = tid << 1;

  float c0 = -INFINITY, c1 = -INFINITY;
  if (tid == 0) c1 = O[(size_t)(SEQ_N - 1) * VOCAB + 1];   // chart[EOS]

  for (int t = SEQ_N - 2; t >= 0; --t) {
    // --- max over chart ---
    red[tid] = fmaxf(c0, c1);
    __syncthreads();
    for (int off = 512; off > 0; off >>= 1) {
      if (tid < off) red[tid] = fmaxf(red[tid], red[tid + off]);
      __syncthreads();
    }
    const float m = red[0];
    // --- w = exp(c - m) ---
    wbuf[r0]     = __expf(c0 - m);
    wbuf[r0 + 1] = __expf(c1 - m);
    __syncthreads();
    // --- s = expT^T w  (coalesced float2 per lane) ---
    float s0 = 0.f, s1 = 0.f;
    for (int j = 0; j < VOCAB; j += 4) {
      const float4 wv = *(const float4*)&wbuf[j];
      const float2 e0 = *(const float2*)(expTt + (size_t)(j + 0) * VOCAB + r0);
      const float2 e1 = *(const float2*)(expTt + (size_t)(j + 1) * VOCAB + r0);
      const float2 e2 = *(const float2*)(expTt + (size_t)(j + 2) * VOCAB + r0);
      const float2 e3 = *(const float2*)(expTt + (size_t)(j + 3) * VOCAB + r0);
      __builtin_prefetch(expTt + (size_t)((j + 64) & (VOCAB - 1)) * VOCAB + r0, 0, 0);
      s0 = fmaf(wv.x, e0.x, fmaf(wv.y, e1.x, fmaf(wv.z, e2.x, fmaf(wv.w, e3.x, s0))));
      s1 = fmaf(wv.x, e0.y, fmaf(wv.y, e1.y, fmaf(wv.z, e2.y, fmaf(wv.w, e3.y, s1))));
    }
    const float2 o = *(const float2*)(O + (size_t)t * VOCAB + r0);
    c0 = o.x + m + __logf(s0);
    c1 = o.y + m + __logf(s1);
    __syncthreads();   // dots done before wbuf/red rewritten next iter
  }
  if (tid == 0) out[0] = c0;   // chart0[BOS], BOS=0 (tid 0 owns rows 0,1)
}

// ---------------------------------------------------------------------------
extern "C" void kernel_launch(void* const* d_in, const int* in_sizes, int n_in,
                              void* d_out, int out_size, void* d_ws, size_t ws_size,
                              hipStream_t stream) {
  (void)in_sizes; (void)n_in; (void)out_size; (void)ws_size;
  const int*   nums = (const int*)d_in[0];
  const float* E    = (const float*)d_in[1];
  const float* Wx1  = (const float*)d_in[2];
  const float* Wh1  = (const float*)d_in[3];
  const float* b1   = (const float*)d_in[4];
  const float* Wx2  = (const float*)d_in[5];
  const float* Wh2  = (const float*)d_in[6];
  const float* b2   = (const float*)d_in[7];
  const float* Wlin = (const float*)d_in[8];
  const float* blin = (const float*)d_in[9];
  const float* T    = (const float*)d_in[10];
  float* out = (float*)d_out;

  char* ws = (char*)d_ws;
  float* XW1   = (float*)ws; ws += sizeof(float) * (size_t)SEQ_N * DIM;
  float* G     = (float*)ws; ws += sizeof(float) * (size_t)SEQ_N * DIM;
  float* XW2   = (float*)ws; ws += sizeof(float) * (size_t)SEQ_N * DIM;
  float* H     = (float*)ws; ws += sizeof(float) * (size_t)SEQ_N * DIM;
  float* O     = (float*)ws; ws += sizeof(float) * (size_t)SEQ_N * VOCAB;
  float* expTt = (float*)ws; ws += sizeof(float) * (size_t)VOCAB * VOCAB;
  uint4* A1p   = (uint4*)ws; ws += 2u * (size_t)SEQ_N * DIM;    // bf16 bytes
  uint4* A2p   = (uint4*)ws; ws += 2u * (size_t)SEQ_N * DIM;
  uint4* A3p   = (uint4*)ws; ws += 2u * (size_t)SEQ_N * DIM;
  uint4* Wx1p  = (uint4*)ws; ws += 2u * (size_t)DIM * DIM;
  uint4* Wx2p  = (uint4*)ws; ws += 2u * (size_t)DIM * DIM;
  uint4* Wlinp = (uint4*)ws; ws += 2u * (size_t)DIM * VOCAB;

  // One-time weight packing into WMMA fragment order (bf16).
  const int bthr1 = (DIM / 16) * (DIM / 32) * 64;     // Wx1 / Wx2
  const int bthr2 = (VOCAB / 16) * (DIM / 32) * 64;   // Wlin
  pack_b_bf16<<<bthr1 / 256, 256, 0, stream>>>(Wx1, Wx1p, DIM, DIM);
  pack_b_bf16<<<bthr1 / 256, 256, 0, stream>>>(Wx2, Wx2p, DIM, DIM);
  pack_b_bf16<<<bthr2 / 256, 256, 0, stream>>>(Wlin, Wlinp, VOCAB, DIM);

  const int athr = (SEQ_N / 16) * (DIM / 32) * 64;
  const int g1blk = (SEQ_N / 16) * (DIM / 64) / 8;     // 16x64 strips, 8 waves/blk
  const int g3blk = (SEQ_N / 16) * (VOCAB / 64) / 8;

  // Encoder: gather-pack -> GEMM -> scan -> pack -> GEMM -> scan -> GEMM.
  pack_a_bf16<<<athr / 256, 256, 0, stream>>>(E, nums, A1p, SEQ_N, DIM);
  wmma_gemm_packed<<<g1blk, 256, 0, stream>>>(A1p, Wx1p, b1, XW1, SEQ_N, DIM, DIM);
  rnn_scan<<<1, 256, 0, stream>>>(XW1, Wh1, G);
  pack_a_bf16<<<athr / 256, 256, 0, stream>>>(G, nullptr, A2p, SEQ_N, DIM);
  wmma_gemm_packed<<<g1blk, 256, 0, stream>>>(A2p, Wx2p, b2, XW2, SEQ_N, DIM, DIM);
  rnn_scan<<<1, 256, 0, stream>>>(XW2, Wh2, H);
  pack_a_bf16<<<athr / 256, 256, 0, stream>>>(H, nullptr, A3p, SEQ_N, DIM);
  wmma_gemm_packed<<<g3blk, 256, 0, stream>>>(A3p, Wlinp, blin, O, SEQ_N, VOCAB, DIM);

  // CRF: exp(T) transposed once, then exp-domain backward scan.
  exp_transpose<<<dim3(VOCAB / 32, VOCAB / 32), dim3(32, 8), 0, stream>>>(T, expTt);
  crf_scan<<<1, 1024, 0, stream>>>(O, expTt, out);
}
